// RelativeScaledDotProductAttention_4750233830026
// MI455X (gfx1250) — matrix operand order repaired and checked by reading, not compiled, not run on hardware
//
#include <hip/hip_runtime.h>

// RelativeScaledDotProductAttention (Music-Transformer skew), B=64 L=1024 D=64.
// out tuple = [output (B,L,D) | out_attn (B,L,L)], fp32.
//
// Roofline: the 268MB out_attn store dominates (~14us @ 23.3TB/s). Matrix work
// runs on bf16 WMMA 16x16x32 with a hi/lo 3-product split (~fp32 accuracy).
// Per-tile f32->bf16 split conversions would add VALU work in the hot loop,
// so K / Er / V are pre-split ONCE into bf16 hi/lo workspace arrays (V also
// transposed for contiguous B-fragment loads); the O(B*L^2) loop then does pure
// fragment loads + WMMA + softmax. Falls back to in-loop conversion if d_ws is
// too small.

#define L_ 1024
#define D_ 64
#define B_ 64
#define NSTRIP 64   // L/16
#define WAVES 8

typedef __attribute__((ext_vector_type(16))) __bf16 v16bf;
typedef __attribute__((ext_vector_type(4)))  __bf16 v4bf;
typedef __attribute__((ext_vector_type(8)))  float  v8f;

__device__ __forceinline__ v8f wmma_bf16(v16bf a, v16bf b, v8f c) {
  return __builtin_amdgcn_wmma_f32_16x16x32_bf16(false, a, false, b, (short)0, c,
                                                 false, false);
}

// split-bf16 matmul-accumulate: (ah+al) x (bh+bl) ~= ah*bh + al*bh + ah*bl
__device__ __forceinline__ v8f mm3(v16bf ah, v16bf al, v16bf bh, v16bf bl, v8f c) {
  c = wmma_bf16(ah, bh, c);
  c = wmma_bf16(al, bh, c);
  c = wmma_bf16(ah, bl, c);
  return c;
}

__device__ __forceinline__ void split1(float x, v16bf& h, v16bf& l, int i) {
  __bf16 a = (__bf16)x;
  h[i] = a;
  l[i] = (__bf16)(x - (float)a);
}

// 16 consecutive f32 -> halfwords h0..h15 (hi/lo)  [fallback path]
__device__ __forceinline__ void cv16(const float* __restrict__ p, v16bf& h, v16bf& l) {
#pragma unroll
  for (int j = 0; j < 4; ++j) {
    float4 x = ((const float4*)p)[j];
    split1(x.x, h, l, 4 * j + 0);
    split1(x.y, h, l, 4 * j + 1);
    split1(x.z, h, l, 4 * j + 2);
    split1(x.w, h, l, 4 * j + 3);
  }
}

// two separated runs of 8 f32 -> h0..7 / h8..15 (A-frag layout, Q only)
__device__ __forceinline__ void cv8x2(const float* __restrict__ pa,
                                      const float* __restrict__ pb,
                                      v16bf& h, v16bf& l) {
#pragma unroll
  for (int j = 0; j < 2; ++j) {
    float4 x = ((const float4*)pa)[j];
    split1(x.x, h, l, 4 * j + 0);
    split1(x.y, h, l, 4 * j + 1);
    split1(x.z, h, l, 4 * j + 2);
    split1(x.w, h, l, 4 * j + 3);
  }
#pragma unroll
  for (int j = 0; j < 2; ++j) {
    float4 x = ((const float4*)pb)[j];
    split1(x.x, h, l, 8 + 4 * j + 0);
    split1(x.y, h, l, 8 + 4 * j + 1);
    split1(x.z, h, l, 8 + 4 * j + 2);
    split1(x.w, h, l, 8 + 4 * j + 3);
  }
}

// strided gather (V columns) -> h0..h15  [fallback path]
__device__ __forceinline__ void cv16s(const float* __restrict__ p, int stride,
                                      v16bf& h, v16bf& l) {
#pragma unroll
  for (int i = 0; i < 16; ++i) split1(p[(size_t)i * stride], h, l, i);
}

// ======================= prep kernels (PRE path) =======================

// flat f32 -> bf16 hi/lo split (K and Er)
__global__ __launch_bounds__(256)
void prep_split(const float* __restrict__ x, __bf16* __restrict__ hi,
                __bf16* __restrict__ lo, int n4) {
  int i = blockIdx.x * blockDim.x + threadIdx.x;
  if (i >= n4) return;
  float4 xv = ((const float4*)x)[i];
  v4bf h, l;
  float e[4] = {xv.x, xv.y, xv.z, xv.w};
#pragma unroll
  for (int j = 0; j < 4; ++j) {
    __bf16 a = (__bf16)e[j];
    h[j] = a;
    l[j] = (__bf16)(e[j] - (float)a);
  }
  ((v4bf*)hi)[i] = h;
  ((v4bf*)lo)[i] = l;
}

// V [B,L,D] f32 -> Vt [B,D,L] bf16 hi/lo (LDS tile transpose, 64x64 tiles)
__global__ __launch_bounds__(256)
void prep_vt(const float* __restrict__ v, __bf16* __restrict__ vthi,
             __bf16* __restrict__ vtlo) {
  __shared__ __bf16 hi[64][66], lo[64][66];   // pad 66 -> conflict-free
  const int b  = blockIdx.x >> 4;
  const int j0 = (blockIdx.x & 15) << 6;
  const int tid = threadIdx.x;
#pragma unroll 4
  for (int i = 0; i < 16; ++i) {
    int idx = tid + 256 * i;
    int r = idx >> 6, d = idx & 63;
    float x = v[((size_t)b * L_ + j0 + r) * D_ + d];
    __bf16 a = (__bf16)x;
    hi[r][d] = a;
    lo[r][d] = (__bf16)(x - (float)a);
  }
  __syncthreads();
#pragma unroll 4
  for (int i = 0; i < 16; ++i) {
    int idx = tid + 256 * i;
    int d = idx >> 6, jj = idx & 63;
    size_t o = ((size_t)b * D_ + d) * L_ + j0 + jj;
    vthi[o] = hi[jj][d];
    vtlo[o] = lo[jj][d];
  }
}

// ========================= main attention kernel =========================

// logits for one 16x16 tile: (Q K^T + skew(Q Er_rev^T)) / 8
template <bool PRE>
__device__ __forceinline__ void tile_logits(
    const float* __restrict__ kbp,
    const __bf16* __restrict__ khi_b, const __bf16* __restrict__ klo_b,
    const float* __restrict__ Er,
    const __bf16* __restrict__ erhi, const __bf16* __restrict__ erlo,
    const v16bf (&QAh)[2], const v16bf (&QAl)[2],
    int qi, int kj, int ln, int half,
    float (*qeL)[32], float (&lg)[8]) {
  v16bf Bh, Bl;
  // content score S = Q K^T  (K-dim = D = 64 -> two 32-chunks)
  v8f sc = {};
#pragma unroll
  for (int c = 0; c < 2; ++c) {
    if constexpr (PRE) {
      const size_t o = (size_t)(kj + ln) * D_ + 32 * c + 16 * half;
      Bh = *(const v16bf*)(khi_b + o);
      Bl = *(const v16bf*)(klo_b + o);
    } else {
      cv16(kbp + (size_t)(kj + ln) * D_ + 32 * c + 16 * half, Bh, Bl);
    }
    sc = mm3(QAh[c], QAl[c], Bh, Bl, sc);
  }
  // relative score: qe2[r][c'] = q_r . Er[L-1-(dbase+c')], dbase = qi-kj-16,
  // then skew gather rel[r][c] = qe2[r][16 + r - c]
  const int dbase = qi - kj - 16;
  v8f qe[2] = {};
#pragma unroll
  for (int t = 0; t < 2; ++t) {
    int d = dbase + 16 * t + ln;
    int e = (L_ - 1) - d;                              // clamped cols are masked
    e = e < 0 ? 0 : (e > (L_ - 1) ? (L_ - 1) : e);
#pragma unroll
    for (int c = 0; c < 2; ++c) {
      if constexpr (PRE) {
        const size_t o = (size_t)e * D_ + 32 * c + 16 * half;
        Bh = *(const v16bf*)(erhi + o);
        Bl = *(const v16bf*)(erlo + o);
      } else {
        cv16(Er + (size_t)e * D_ + 32 * c + 16 * half, Bh, Bl);
      }
      qe[t] = mm3(QAh[c], QAl[c], Bh, Bl, qe[t]);
    }
  }
  // stage 16x32 qe slab in LDS, then skewed gather
#pragma unroll
  for (int t = 0; t < 2; ++t)
#pragma unroll
    for (int g = 0; g < 8; ++g)
      qeL[g + 8 * half][16 * t + ln] = qe[t][g];
  asm volatile("s_wait_dscnt 0x0" ::: "memory");
#pragma unroll
  for (int g = 0; g < 8; ++g) {
    int r = g + 8 * half;
    float rel = qeL[r][16 + r - ln];
    lg[g] = (sc[g] + rel) * 0.125f;
  }
}

template <bool PRE>
__global__ __launch_bounds__(256, 1)
void relattn_fused(const float* __restrict__ q, const float* __restrict__ k,
                   const float* __restrict__ v, const float* __restrict__ Er,
                   const __bf16* __restrict__ khi, const __bf16* __restrict__ klo,
                   const __bf16* __restrict__ vthi, const __bf16* __restrict__ vtlo,
                   const __bf16* __restrict__ erhi, const __bf16* __restrict__ erlo,
                   float* __restrict__ out, float* __restrict__ attn) {
  __shared__ float qe_buf[WAVES][16][32];
  __shared__ float p_buf[WAVES][16][32];

  const int tid  = threadIdx.x;
  const int wv   = tid >> 5;
  const int lane = tid & 31;
  const int half = lane >> 4;
  const int ln   = lane & 15;

  const int w  = blockIdx.x * WAVES + wv;   // global wave id
  const int b  = w >> 6;                    // batch
  const int s  = w & (NSTRIP - 1);          // query strip index
  const int qi = s << 4;

  const float* qb = q + (size_t)b * L_ * D_;
  const float* kb = k + (size_t)b * L_ * D_;
  const float* vb = v + (size_t)b * L_ * D_;
  const __bf16* khi_b = khi + (size_t)b * L_ * D_;
  const __bf16* klo_b = klo + (size_t)b * L_ * D_;
  float* attnb = attn + (size_t)b * L_ * L_;

  // persistent Q A-fragments (two K-chunks, hi/lo) - converted once per wave
  v16bf QAh[2], QAl[2];
#pragma unroll
  for (int c = 0; c < 2; ++c) {
    const float* qr = qb + (size_t)(qi + ln) * D_ + 32 * c + 16 * half;
    cv8x2(qr, qr + 8, QAh[c], QAl[c]);
  }

  // ---------------- pass 1: per-row sum of exp(logit) ----------------
  float rinv[8];
#pragma unroll
  for (int g = 0; g < 8; ++g) rinv[g] = 0.f;

  for (int kt = 0; kt <= s; ++kt) {
    const int kj = kt << 4;
    if (kt < s) {
      if constexpr (PRE)
        __builtin_prefetch(khi_b + (size_t)(kj + 16 + ln) * D_, 0, 1);
      else
        __builtin_prefetch(kb + (size_t)(kj + 16 + ln) * D_, 0, 1);
    }
    float lg[8];
    tile_logits<PRE>(kb, khi_b, klo_b, Er, erhi, erlo, QAh, QAl,
                     qi, kj, ln, half, qe_buf[wv], lg);
#pragma unroll
    for (int g = 0; g < 8; ++g) {
      int r = g + 8 * half;
      float p = __expf(lg[g]);
      p = (kt == s && ln > r) ? 0.f : p;   // causal mask on diagonal tile
      rinv[g] += p;
    }
  }
  // row reduction across the 16 lanes of each half, then invert
#pragma unroll
  for (int g = 0; g < 8; ++g) {
#pragma unroll
    for (int m = 1; m < 16; m <<= 1) rinv[g] += __shfl_xor(rinv[g], m, 32);
    rinv[g] = 1.0f / rinv[g];
  }

  // ------------- pass 2: write probs once + accumulate P.V -------------
  v8f Oa[4] = {};
  for (int kt = 0; kt <= s; ++kt) {
    const int kj = kt << 4;
    float lg[8];
    tile_logits<PRE>(kb, khi_b, klo_b, Er, erhi, erlo, QAh, QAl,
                     qi, kj, ln, half, qe_buf[wv], lg);
    const int ph = (kt & 1) << 4;
#pragma unroll
    for (int g = 0; g < 8; ++g) {
      int r = g + 8 * half;
      float p = __expf(lg[g]) * rinv[g];
      p = (kt == s && ln > r) ? 0.f : p;
      attnb[(size_t)(qi + r) * L_ + kj + ln] = p;   // single write of attn
      p_buf[wv][r][ph + ln] = p;                    // stage for PV transpose
    }
    if ((kt & 1) || kt == s) {                      // flush 32-key PV block
      if (!(kt & 1)) {                              // odd count: zero-pad half
#pragma unroll
        for (int g = 0; g < 8; ++g) p_buf[wv][g + 8 * half][16 + ln] = 0.f;
      }
      asm volatile("s_wait_dscnt 0x0" ::: "memory");
      v16bf Ph, Pl;
#pragma unroll
      for (int i = 0; i < 8; ++i) {                 // P as A-frag (K=32 keys)
        split1(p_buf[wv][ln][16 * half + i],     Ph, Pl, i);
        split1(p_buf[wv][ln][16 * half + 8 + i], Ph, Pl, 8 + i);
      }
      const int kb0 = (kt & ~1) << 4;
      v16bf Vh, Vl;
#pragma unroll
      for (int n = 0; n < 4; ++n) {                 // four 16-wide D tiles
        if constexpr (PRE) {
          const size_t o = ((size_t)b * D_ + 16 * n + ln) * L_ + kb0 + 16 * half;
          Vh = *(const v16bf*)(vthi + o);
          Vl = *(const v16bf*)(vtlo + o);
        } else {
          cv16s(vb + (size_t)(kb0 + 16 * half) * D_ + 16 * n + ln, D_, Vh, Vl);
        }
        Oa[n] = mm3(Ph, Pl, Vh, Vl, Oa[n]);
      }
    }
  }

  // write output tile (16 x 64)
#pragma unroll
  for (int n = 0; n < 4; ++n)
#pragma unroll
    for (int g = 0; g < 8; ++g)
      out[(size_t)(b * L_ + qi + g + 8 * half) * D_ + 16 * n + ln] = Oa[n][g];

  // masked (strictly-upper) attn region is exact zero in the reference
  const int c40 = ((s + 1) << 4) >> 2;
  const float4 z4 = make_float4(0.f, 0.f, 0.f, 0.f);
  for (int r = 0; r < 16; ++r) {
    float4* row4 = (float4*)(attnb + (size_t)(qi + r) * L_);
    for (int c4 = c40 + lane; c4 < (L_ >> 2); c4 += 32) row4[c4] = z4;
  }
}

extern "C" void kernel_launch(void* const* d_in, const int* in_sizes, int n_in,
                              void* d_out, int out_size, void* d_ws, size_t ws_size,
                              hipStream_t stream) {
  (void)in_sizes; (void)n_in; (void)out_size;
  const float* q  = (const float*)d_in[0];
  const float* k  = (const float*)d_in[1];
  const float* v  = (const float*)d_in[2];
  const float* Er = (const float*)d_in[3];
  // d_in[4] is the causal mask; causality is hardcoded in the kernel.
  float* out  = (float*)d_out;
  float* attn = out + (size_t)B_ * L_ * D_;   // outputs concatenated flat

  const size_t NKV = (size_t)B_ * L_ * D_;    // 4,194,304 elements
  const size_t NER = (size_t)L_ * D_;         // 65,536 elements
  const size_t need = (4 * NKV + 2 * NER) * sizeof(__bf16);  // ~33.8 MB

  dim3 block(WAVES * 32);
  dim3 grid(B_ * NSTRIP / WAVES);

  if (ws_size >= need) {
    __bf16* khi  = (__bf16*)d_ws;
    __bf16* klo  = khi + NKV;
    __bf16* vthi = klo + NKV;
    __bf16* vtlo = vthi + NKV;
    __bf16* erhi = vtlo + NKV;
    __bf16* erlo = erhi + NER;
    prep_split<<<dim3((unsigned)((NKV / 4 + 255) / 256)), block, 0, stream>>>(
        k, khi, klo, (int)(NKV / 4));
    prep_split<<<dim3((unsigned)((NER / 4 + 255) / 256)), block, 0, stream>>>(
        Er, erhi, erlo, (int)(NER / 4));
    prep_vt<<<dim3(B_ * 16), block, 0, stream>>>(v, vthi, vtlo);
    relattn_fused<true><<<grid, block, 0, stream>>>(
        q, k, v, Er, khi, klo, vthi, vtlo, erhi, erlo, out, attn);
  } else {
    relattn_fused<false><<<grid, block, 0, stream>>>(
        q, k, v, Er, nullptr, nullptr, nullptr, nullptr, nullptr, nullptr,
        out, attn);
  }
}
